// NPMaskChangeTransitionPrior_39075612459510
// MI455X (gfx1250) — matrix operand order repaired and checked by reading, not compiled, not run on hardware
//
#include <hip/hip_runtime.h>
#include <math.h>

// ---------------------------------------------------------------- constants
#define BB     128
#define TT     2050
#define NWIN   2048                    // TT - LAGS
#define DD     8
#define NTILES (BB * NWIN / 16)        // 16384 window tiles (16 windows each)
#define RROWS  (BB * TT)               // 262400 time rows
#define RTILES (RROWS / 16)            // 16400

typedef __attribute__((ext_vector_type(16))) _Float16 v16h;
typedef __attribute__((ext_vector_type(8)))  float    v8f;

union F16x16 { v16h v; uint4 q[2]; _Float16 h[16]; };

__device__ __forceinline__ v8f zero8() {
  v8f z = {0.f, 0.f, 0.f, 0.f, 0.f, 0.f, 0.f, 0.f};
  return z;
}

__device__ __forceinline__ v8f wmma32(const F16x16& a, const F16x16& b, v8f c) {
  // D(16x16,f32) = A(16x32,f16) * B(32x16,f16) + C
  return __builtin_amdgcn_wmma_f32_16x16x32_f16(false, a.v, false, b.v,
                                                (short)0, c, false, false);
}

// B-fragment stored as [frag][lane][16 halfs] contiguous -> one 32B read/lane.
__device__ __forceinline__ F16x16 ld_frag(const _Float16* p, int lane) {
  F16x16 r;
  const uint4* q = (const uint4*)(p + lane * 16);
  r.q[0] = q[0]; r.q[1] = q[1];
  return r;
}

// Read one 16x32 A fragment (chunk kc) from a row-major 16x64 f16 staging tile.
// A layout: lane (m=l%16, g=l/16), half j -> K = kc*32 + 8*g + (j%8) + 16*(j/8)
// => two contiguous 16B reads per lane.
__device__ __forceinline__ F16x16 read_afrag(const _Float16* stage, int kc, int lane) {
  const int m = lane & 15, g = lane >> 4;
  F16x16 r;
  r.q[0] = *(const uint4*)(stage + m * 64 + kc * 32 + g * 8);
  r.q[1] = *(const uint4*)(stage + m * 64 + kc * 32 + 16 + g * 8);
  return r;
}

// Convert KxN f32 weights -> f16 B-fragments in LDS (zero padding outside
// Kreal/Nreal).  Fragment id = kc*NC + nc; element: lane (n=l%16,g=l/16),
// half j = W[kc*32 + g*8 + (j%8) + 16*(j/8)][nc*16 + n].
__device__ __forceinline__ void load_bfrags(_Float16* dst, const float* W,
                                            int Kreal, int Nreal, int NC, int KC,
                                            int ldw, int tid) {
  const int total = KC * NC * 512;
  for (int e = tid; e < total; e += 32) {
    int j    = e & 15;
    int lane = (e >> 4) & 31;
    int frag = e >> 9;
    int kc = frag / NC, nc = frag % NC;
    int g = lane >> 4, nl = lane & 15;
    int k = kc * 32 + g * 8 + (j & 7) + 16 * (j >> 3);
    int n = nc * 16 + nl;
    float v = (k < Kreal && n < Nreal) ? W[(size_t)k * ldw + n] : 0.0f;
    dst[e] = (_Float16)v;
  }
}

// ---------------------------------------------------------------- kZ: zero sld
__global__ void __launch_bounds__(128) kZ(float* sld) {
  if (threadIdx.x < BB) sld[threadIdx.x] = 0.0f;
}

// ---------------------------------------------------------------- kA: H = nn MLP(x, x@mask)
__global__ void __launch_bounds__(32)
kA(const float* __restrict__ x, const float* __restrict__ u_noise,
   const float* __restrict__ W0, const float* __restrict__ b0,
   const float* __restrict__ W1, const float* __restrict__ b1,
   const float* __restrict__ W2, const float* __restrict__ b2,
   _Float16* __restrict__ Hout) {
  __shared__ __align__(16) _Float16 sW0[4  * 512];  // 16x64, K padded to 32
  __shared__ __align__(16) _Float16 sW1[8  * 512];  // 64x64
  __shared__ __align__(16) _Float16 sW2[2  * 512];  // 64x8, N padded to 16
  __shared__ float sB0[64], sB1[64], sB2m[8], sMask[64];
  __shared__ __align__(16) _Float16 stg[16 * 64];
  __shared__ __align__(16) _Float16 stO[16 * 8];

  const int tid = threadIdx.x;
  load_bfrags(sW0, W0, 16, 64, 4, 1, 64, tid);
  load_bfrags(sW1, W1, 64, 64, 4, 2, 64, tid);
  load_bfrags(sW2, W2, 64,  8, 1, 2,  8, tid);
  for (int i = tid; i < 64; i += 32) { sB0[i] = b0[i]; sB1[i] = b1[i]; }
  if (tid < 8) sB2m[tid] = b2[tid];
  for (int e = tid; e < 64; e += 32) {         // gumbel-sigmoid mask, zero diag
    int i = e >> 3, j = e & 7;
    float u = u_noise[e];
    u = fminf(fmaxf(u, 1e-6f), 1.0f - 1e-6f);
    float lg = (logf(u) - log1pf(-u)) * (1.0f / 0.3f);
    float mv = 1.0f / (1.0f + expf(-lg));
    sMask[e] = (i == j) ? 0.0f : mv;
  }
  __syncthreads();

  const int lane = tid, nl = lane & 15, g = lane >> 4;
  for (int t = blockIdx.x; t < RTILES; t += gridDim.x) {
    const int row = t * 16 + nl;
    float xv[8];
    const float* xr = x + (size_t)row * 8;
#pragma unroll
    for (int j = 0; j < 8; ++j) xv[j] = xr[j];
    F16x16 A;                                   // K: [x(8) | xm(8) | pad]
    if (g == 0) {
#pragma unroll
      for (int j = 0; j < 8; ++j) A.h[j] = (_Float16)xv[j];
    } else {
#pragma unroll
      for (int j = 0; j < 8; ++j) {
        float s = 0.f;
#pragma unroll
        for (int i2 = 0; i2 < 8; ++i2) s += xv[i2] * sMask[i2 * 8 + j];
        A.h[j] = (_Float16)s;
      }
    }
#pragma unroll
    for (int j = 8; j < 16; ++j) A.h[j] = (_Float16)0.0f;

    v8f z[4];
#pragma unroll
    for (int nc = 0; nc < 4; ++nc) z[nc] = wmma32(A, ld_frag(sW0 + nc * 512, lane), zero8());
#pragma unroll
    for (int nc = 0; nc < 4; ++nc) {
      float bv = sB0[nc * 16 + nl];
#pragma unroll
      for (int r = 0; r < 8; ++r) {
        float zz = z[nc][r] + bv;
        stg[(r + 8 * g) * 64 + nc * 16 + nl] = (_Float16)(zz >= 0.f ? zz : 0.2f * zz);
      }
    }
    __syncthreads();
    F16x16 hA0 = read_afrag(stg, 0, lane), hA1 = read_afrag(stg, 1, lane);
    __syncthreads();

#pragma unroll
    for (int nc = 0; nc < 4; ++nc) {
      v8f a = zero8();
      a = wmma32(hA0, ld_frag(sW1 + nc * 512, lane), a);
      a = wmma32(hA1, ld_frag(sW1 + (4 + nc) * 512, lane), a);
      z[nc] = a;
    }
#pragma unroll
    for (int nc = 0; nc < 4; ++nc) {
      float bv = sB1[nc * 16 + nl];
#pragma unroll
      for (int r = 0; r < 8; ++r) {
        float zz = z[nc][r] + bv;
        stg[(r + 8 * g) * 64 + nc * 16 + nl] = (_Float16)(zz >= 0.f ? zz : 0.2f * zz);
      }
    }
    __syncthreads();
    hA0 = read_afrag(stg, 0, lane); hA1 = read_afrag(stg, 1, lane);
    __syncthreads();

    v8f zo = zero8();                                   // last layer, linear
    zo = wmma32(hA0, ld_frag(sW2 + 0 * 512, lane), zo);
    zo = wmma32(hA1, ld_frag(sW2 + 1 * 512, lane), zo);
    if (nl < 8) {
      float bv = sB2m[nl];
#pragma unroll
      for (int r = 0; r < 8; ++r) stO[(r + 8 * g) * 8 + nl] = (_Float16)(zo[r] + bv);
    }
    __syncthreads();
    if (g == 0) *(uint4*)(Hout + (size_t)row * 8) = *(const uint4*)(stO + nl * 8);
    __syncthreads();
  }
}

// ---------------------------------------------------------------- kB: emb_h, stored A-swizzled
__global__ void __launch_bounds__(32)
kB(const float* __restrict__ emb,
   const float* __restrict__ W0, const float* __restrict__ b0,
   const float* __restrict__ W1, const float* __restrict__ b1,
   const float* __restrict__ W2, const float* __restrict__ b2,
   _Float16* __restrict__ wsEmb) {
  __shared__ __align__(16) _Float16 sW0[4 * 512];   // 8x64, K padded to 32
  __shared__ __align__(16) _Float16 sW1[8 * 512];
  __shared__ __align__(16) _Float16 sW2[8 * 512];
  __shared__ float sB0[64], sB1[64], sB2[64];
  __shared__ __align__(16) _Float16 stg[16 * 64];

  const int tid = threadIdx.x;
  load_bfrags(sW0, W0,  8, 64, 4, 1, 64, tid);
  load_bfrags(sW1, W1, 64, 64, 4, 2, 64, tid);
  load_bfrags(sW2, W2, 64, 64, 4, 2, 64, tid);
  for (int i = tid; i < 64; i += 32) { sB0[i] = b0[i]; sB1[i] = b1[i]; sB2[i] = b2[i]; }
  __syncthreads();

  const int lane = tid, nl = lane & 15, g = lane >> 4;
  for (int t = blockIdx.x; t < NTILES; t += gridDim.x) {
    F16x16 A;
#pragma unroll
    for (int j = 0; j < 16; ++j) A.h[j] = (_Float16)0.0f;
    if (g == 0) {                                      // K=0..7 = embedding row
      const float* er = emb + (size_t)(t * 16 + nl) * 8;
#pragma unroll
      for (int j = 0; j < 8; ++j) A.h[j] = (_Float16)er[j];
    }

    v8f z[4];
#pragma unroll
    for (int nc = 0; nc < 4; ++nc) z[nc] = wmma32(A, ld_frag(sW0 + nc * 512, lane), zero8());
#pragma unroll
    for (int nc = 0; nc < 4; ++nc) {
      float bv = sB0[nc * 16 + nl];
#pragma unroll
      for (int r = 0; r < 8; ++r) {
        float zz = z[nc][r] + bv;
        stg[(r + 8 * g) * 64 + nc * 16 + nl] = (_Float16)(zz >= 0.f ? zz : 0.2f * zz);
      }
    }
    __syncthreads();
    F16x16 hA0 = read_afrag(stg, 0, lane), hA1 = read_afrag(stg, 1, lane);
    __syncthreads();

#pragma unroll
    for (int nc = 0; nc < 4; ++nc) {
      v8f a = zero8();
      a = wmma32(hA0, ld_frag(sW1 + nc * 512, lane), a);
      a = wmma32(hA1, ld_frag(sW1 + (4 + nc) * 512, lane), a);
      z[nc] = a;
    }
#pragma unroll
    for (int nc = 0; nc < 4; ++nc) {
      float bv = sB1[nc * 16 + nl];
#pragma unroll
      for (int r = 0; r < 8; ++r) {
        float zz = z[nc][r] + bv;
        stg[(r + 8 * g) * 64 + nc * 16 + nl] = (_Float16)(zz >= 0.f ? zz : 0.2f * zz);
      }
    }
    __syncthreads();
    hA0 = read_afrag(stg, 0, lane); hA1 = read_afrag(stg, 1, lane);
    __syncthreads();

#pragma unroll
    for (int nc = 0; nc < 4; ++nc) {
      v8f a = zero8();
      a = wmma32(hA0, ld_frag(sW2 + nc * 512, lane), a);
      a = wmma32(hA1, ld_frag(sW2 + (4 + nc) * 512, lane), a);
      z[nc] = a;
    }
#pragma unroll
    for (int nc = 0; nc < 4; ++nc) {                   // linear last layer
      float bv = sB2[nc * 16 + nl];
#pragma unroll
      for (int r = 0; r < 8; ++r)
        stg[(r + 8 * g) * 64 + nc * 16 + nl] = (_Float16)(z[nc][r] + bv);
    }
    __syncthreads();
    F16x16 c0 = read_afrag(stg, 0, lane), c1 = read_afrag(stg, 1, lane);
    uint4* d0 = (uint4*)(wsEmb + ((size_t)(t * 2 + 0) * 32 + lane) * 16);
    d0[0] = c0.q[0]; d0[1] = c0.q[1];
    uint4* d1 = (uint4*)(wsEmb + ((size_t)(t * 2 + 1) * 32 + lane) * 16);
    d1[0] = c1.q[0]; d1[1] = c1.q[1];
    __syncthreads();
  }
}

// ---------------------------------------------------------------- kC helpers
__device__ __forceinline__ void gs_mid_layer(const _Float16* sW, const float* sB,
                                             F16x16& hA0, F16x16& hA1,
                                             F16x16& tA0, F16x16& tA1,
                                             _Float16* stgH, _Float16* stgT, int lane) {
  const int nl = lane & 15, g = lane >> 4;
  v8f z[4], tz[4];
#pragma unroll
  for (int nc = 0; nc < 4; ++nc) {
    F16x16 bf0 = ld_frag(sW + nc * 512, lane);
    F16x16 bf1 = ld_frag(sW + (4 + nc) * 512, lane);
    v8f a = zero8(); a = wmma32(hA0, bf0, a); a = wmma32(hA1, bf1, a); z[nc]  = a;
    v8f b = zero8(); b = wmma32(tA0, bf0, b); b = wmma32(tA1, bf1, b); tz[nc] = b;
  }
#pragma unroll
  for (int nc = 0; nc < 4; ++nc) {
    float bv = sB[nc * 16 + nl];
#pragma unroll
    for (int r = 0; r < 8; ++r) {
      float zz = z[nc][r] + bv;
      float hv = zz >= 0.f ? zz : 0.2f * zz;
      float dv = zz >= 0.f ? 1.0f : 0.2f;
      int ad = (r + 8 * g) * 64 + nc * 16 + nl;
      stgH[ad] = (_Float16)hv;
      stgT[ad] = (_Float16)(dv * tz[nc][r]);
    }
  }
  __syncthreads();
  hA0 = read_afrag(stgH, 0, lane); hA1 = read_afrag(stgH, 1, lane);
  tA0 = read_afrag(stgT, 0, lane); tA1 = read_afrag(stgT, 1, lane);
  __syncthreads();
}

// ---------------------------------------------------------------- kC: gs MLP + JVP per d
__global__ void __launch_bounds__(32)
kC(const _Float16* __restrict__ Hws, const _Float16* __restrict__ wsEmb,
   const float* __restrict__ gsW0, const float* __restrict__ gsB0,
   const float* __restrict__ gsW1, const float* __restrict__ gsB1,
   const float* __restrict__ gsW2, const float* __restrict__ gsB2,
   const float* __restrict__ gsW3, const float* __restrict__ gsB3,
   float* __restrict__ outRes, float* __restrict__ outSld) {
  __shared__ __align__(16) _Float16 sW0[12 * 512];  // 81x64, K padded to 96
  __shared__ __align__(16) _Float16 sW1[8  * 512];
  __shared__ __align__(16) _Float16 sW2[8  * 512];
  __shared__ __align__(16) _Float16 sW3[2  * 512];  // 64x1, N padded to 16
  __shared__ float sB0[64], sB1[64], sB2[64], sT0[64];
  __shared__ float sB3s;
  __shared__ __align__(16) _Float16 stgH[16 * 64];
  __shared__ __align__(16) _Float16 stgT[16 * 64];

  const int tid = threadIdx.x;
  const int d   = blockIdx.x;                         // one dim-network per block
  load_bfrags(sW0, gsW0 + (size_t)d * 81 * 64, 81, 64, 4, 3, 64, tid);
  load_bfrags(sW1, gsW1 + (size_t)d * 64 * 64, 64, 64, 4, 2, 64, tid);
  load_bfrags(sW2, gsW2 + (size_t)d * 64 * 64, 64, 64, 4, 2, 64, tid);
  load_bfrags(sW3, gsW3 + (size_t)d * 64,      64,  1, 1, 2,  1, tid);
  for (int i = tid; i < 64; i += 32) {
    sB0[i] = gsB0[d * 64 + i];
    sB1[i] = gsB1[d * 64 + i];
    sB2[i] = gsB2[d * 64 + i];
    sT0[i] = gsW0[((size_t)d * 81 + 80) * 64 + i];    // tangent seed = W0 row 80
  }
  if (tid == 0) sB3s = gsB3[d];
  __syncthreads();

  const int lane = tid, nl = lane & 15, g = lane >> 4;
  for (int t = blockIdx.y; t < NTILES; t += gridDim.y) {
    const int bb = t >> 7;                            // 128 tiles per batch row
    const int w0 = (t & 127) << 4;

    // A chunks: [emb_h(64) | yy(16) xx(1) pad(15)]
    F16x16 A0, A1, A2;
    const uint4* p0 = (const uint4*)(wsEmb + ((size_t)(t * 2 + 0) * 32 + lane) * 16);
    A0.q[0] = p0[0]; A0.q[1] = p0[1];
    const uint4* p1 = (const uint4*)(wsEmb + ((size_t)(t * 2 + 1) * 32 + lane) * 16);
    A1.q[0] = p1[0]; A1.q[1] = p1[1];
    {
      const int w = w0 + nl;
      // g=0 -> K 64..71 = H[b,w]; g=1 -> K 72..79 = H[b,w+1]
      A2.q[0] = *(const uint4*)(Hws + ((size_t)bb * TT + w + g) * 8);
#pragma unroll
      for (int j = 8; j < 16; ++j) A2.h[j] = (_Float16)0.0f;
      if (g == 0)                                      // K=80: xx_d = H[b,w+2][d]
        A2.h[8] = Hws[((size_t)bb * TT + w + 2) * 8 + d];
    }

    // layer 0 (K=96): forward only; tangent seeded from sT0
    v8f z[4];
#pragma unroll
    for (int nc = 0; nc < 4; ++nc) {
      v8f a = zero8();
      a = wmma32(A0, ld_frag(sW0 + (0 * 4 + nc) * 512, lane), a);
      a = wmma32(A1, ld_frag(sW0 + (1 * 4 + nc) * 512, lane), a);
      a = wmma32(A2, ld_frag(sW0 + (2 * 4 + nc) * 512, lane), a);
      z[nc] = a;
    }
    F16x16 hA0, hA1, tA0, tA1;
#pragma unroll
    for (int nc = 0; nc < 4; ++nc) {
      const int col = nc * 16 + nl;
      float bv = sB0[col], ts = sT0[col];
#pragma unroll
      for (int r = 0; r < 8; ++r) {
        float zz = z[nc][r] + bv;
        float hv = zz >= 0.f ? zz : 0.2f * zz;
        float dv = zz >= 0.f ? 1.0f : 0.2f;
        int ad = (r + 8 * g) * 64 + col;
        stgH[ad] = (_Float16)hv;
        stgT[ad] = (_Float16)(dv * ts);
      }
    }
    __syncthreads();
    hA0 = read_afrag(stgH, 0, lane); hA1 = read_afrag(stgH, 1, lane);
    tA0 = read_afrag(stgT, 0, lane); tA1 = read_afrag(stgT, 1, lane);
    __syncthreads();

    gs_mid_layer(sW1, sB1, hA0, hA1, tA0, tA1, stgH, stgT, lane);
    gs_mid_layer(sW2, sB2, hA0, hA1, tA0, tA1, stgH, stgT, lane);

    // output layer: W3 padded to 64x16 (only column 0 non-zero)
    v8f res = zero8(), pd = zero8();
    {
      F16x16 bf0 = ld_frag(sW3 + 0 * 512, lane);
      F16x16 bf1 = ld_frag(sW3 + 1 * 512, lane);
      res = wmma32(hA0, bf0, res); res = wmma32(hA1, bf1, res);
      pd  = wmma32(tA0, bf0, pd);  pd  = wmma32(tA1, bf1, pd);
    }
    if (nl == 0) {                                     // lanes 0,16 own column 0
      float lgs = 0.f;
#pragma unroll
      for (int r = 0; r < 8; ++r) {
        const int w = w0 + r + 8 * g;
        outRes[((size_t)bb * NWIN + w) * 8 + d] = res[r] + sB3s;
        lgs += logf(fabsf(pd[r]));
      }
      float oth = __shfl_xor(lgs, 16);
      if (lane == 0) atomicAdd(outSld + bb, lgs + oth);
    }
  }
}

// ---------------------------------------------------------------- host entry
extern "C" void kernel_launch(void* const* d_in, const int* in_sizes, int n_in,
                              void* d_out, int out_size, void* d_ws, size_t ws_size,
                              hipStream_t stream) {
  const float* x     = (const float*)d_in[0];
  const float* emb   = (const float*)d_in[1];
  const float* unz   = (const float*)d_in[2];
  const float* fcW0  = (const float*)d_in[3];
  const float* fcb0  = (const float*)d_in[4];
  const float* fcW1  = (const float*)d_in[5];
  const float* fcb1  = (const float*)d_in[6];
  const float* fcW2  = (const float*)d_in[7];
  const float* fcb2  = (const float*)d_in[8];
  const float* nnW0  = (const float*)d_in[9];
  const float* nnb0  = (const float*)d_in[10];
  const float* nnW1  = (const float*)d_in[11];
  const float* nnb1  = (const float*)d_in[12];
  const float* nnW2  = (const float*)d_in[13];
  const float* nnb2  = (const float*)d_in[14];
  const float* gsW0  = (const float*)d_in[15];
  const float* gsb0  = (const float*)d_in[16];
  const float* gsW1  = (const float*)d_in[17];
  const float* gsb1  = (const float*)d_in[18];
  const float* gsW2  = (const float*)d_in[19];
  const float* gsb2  = (const float*)d_in[20];
  const float* gsW3  = (const float*)d_in[21];
  const float* gsb3  = (const float*)d_in[22];

  float* outRes = (float*)d_out;                        // (B, NWIN, D)
  float* outSld = outRes + (size_t)BB * NWIN * DD;      // (B,)

  // workspace: H (B*T*8 f16) then emb_h A-fragments (NTILES*2*512 f16)
  _Float16* Hws   = (_Float16*)d_ws;
  _Float16* wsEmb = Hws + (size_t)RROWS * 8;
  size_t need = ((size_t)RROWS * 8 + (size_t)NTILES * 2 * 512) * sizeof(_Float16);
  if (ws_size < need) return;

  kZ<<<1, 128, 0, stream>>>(outSld);
  kA<<<2048, 32, 0, stream>>>(x, unz, nnW0, nnb0, nnW1, nnb1, nnW2, nnb2, Hws);
  kB<<<2048, 32, 0, stream>>>(emb, fcW0, fcb0, fcW1, fcb1, fcW2, fcb2, wsEmb);
  kC<<<dim3(DD, 1024), 32, 0, stream>>>(Hws, wsEmb,
                                        gsW0, gsb0, gsW1, gsb1,
                                        gsW2, gsb2, gsW3, gsb3,
                                        outRes, outSld);
}
